// TriangleMultiplication_44006234914906
// MI455X (gfx1250) — matrix operand order repaired and checked by reading, not compiled, not run on hardware
//
#include <hip/hip_runtime.h>
#include <hip/hip_bf16.h>

#define N_SEQ 384
#define C_DIM 128
#define H_DIM 128
#define LDH   130   // padded LDS row stride in halves
#define LDW   65    // same stride in dwords

typedef __attribute__((ext_vector_type(16))) _Float16 v16h;
typedef __attribute__((ext_vector_type(8)))  float    v8f;

union FragH { v16h v; uint32_t u[8]; };

// A-matrix 16x32 f16 fragment per ISA 7.12.2:
// lane L (0..15): M=L, VGPR j<4 -> K=2j,2j+1 ; j>=4 -> K=16+2(j-4)..
// lane L (16..31): M=L-16, K shifted by +8.
__device__ __forceinline__ v16h load_frag_A(const uint32_t* s, int m0, int K0, int lane) {
    FragH f;
    const int m  = m0 + (lane & 15);
    const int hi = (lane >> 4) * 8;
    const uint32_t* row = s + m * LDW;
#pragma unroll
    for (int j = 0; j < 8; ++j) {
        const int K = K0 + ((j & 4) << 2) + hi + 2 * (j & 3);
        f.u[j] = row[K >> 1];
    }
    return f.v;
}

// B-matrix 32x16 f16 fragment; weight stored transposed in LDS as [n][k].
// lane L: N=L&15, K order same as A: VGPR j<4 -> K=2j,2j+1 ; j>=4 -> K=16+2(j-4).., +8 for L>=16.
__device__ __forceinline__ v16h load_frag_B(const uint32_t* s, int n0, int K0, int lane) {
    FragH f;
    const int n = n0 + (lane & 15);
    const uint32_t* row = s + n * LDW;
#pragma unroll
    for (int j = 0; j < 8; ++j) f.u[j] = row[(K0 + ((j & 4) << 2) + ((lane >> 4) * 8) + 2 * (j & 3)) >> 1];
    return f.v;
}

// ---- prep: transposed f16 weights + combined bias --------------------------
__global__ void prep_weights(const float* __restrict__ Wl, const float* __restrict__ We,
                             const float* __restrict__ Wg, const float* __restrict__ Wo,
                             const float* __restrict__ bl, const float* __restrict__ be,
                             _Float16* __restrict__ WTle, _Float16* __restrict__ WTg,
                             _Float16* __restrict__ WTo, float* __restrict__ ble) {
    const int idx = blockIdx.x * blockDim.x + threadIdx.x;   // 0..49151
    const int which = idx >> 14;
    const int r = idx & 16383;
    const int a = r >> 7, b = r & 127;
    if (which == 0)      WTle[r] = (_Float16)(Wl[b * H_DIM + a] + We[b * H_DIM + a]); // [h][c]
    else if (which == 1) WTg[r]  = (_Float16)Wg[b * H_DIM + a];                       // [h][c]
    else                 WTo[r]  = (_Float16)Wo[b * C_DIM + a];                       // [c][h]
    if (idx < 128) ble[idx] = bl[idx] + be[idx];
}

// ---- rs[n][h] = colsum_j(pair[n,j,:]) @ W_right + N*b_right ----------------
__global__ void row_sums(const float* __restrict__ pair, const float* __restrict__ Wr,
                         const float* __restrict__ br, float* __restrict__ rs) {
    __shared__ float cs[C_DIM];
    const int n = blockIdx.x, t = threadIdx.x;
    const float* p = pair + (size_t)n * N_SEQ * C_DIM;
    float a = 0.f;
    for (int j = 0; j < N_SEQ; ++j) a += p[(size_t)j * C_DIM + t];
    cs[t] = a;
    __syncthreads();
    float acc = (float)N_SEQ * br[t];
    for (int c = 0; c < C_DIM; ++c) acc += cs[c] * Wr[c * H_DIM + t];
    rs[n * H_DIM + t] = acc;
}

// ---- main fused kernel: one WG per (n, 128-row chunk) ----------------------
__global__ __launch_bounds__(256)
void tri_main(const float* __restrict__ pair, const float* __restrict__ rs_g,
              const _Float16* __restrict__ WTle_g, const _Float16* __restrict__ WTg_g,
              const _Float16* __restrict__ WTo_g,
              const float* __restrict__ ble_g, const float* __restrict__ bg_g,
              const float* __restrict__ bo_g, const float* __restrict__ lnsc,
              const float* __restrict__ lnb, float* __restrict__ out) {
    extern __shared__ char smem[];
    uint32_t* sWle = (uint32_t*)smem;              // phase1; reused for W_out phase2
    uint32_t* sWg  = (uint32_t*)(smem + 33280);
    uint32_t* sA   = (uint32_t*)(smem + 66560);
    uint32_t* sT   = (uint32_t*)(smem + 99840);
    float*    rsL  = (float*)(smem + 133120);
    _Float16* sA_h = (_Float16*)sA;
    _Float16* sT_h = (_Float16*)sT;

    const int tid  = threadIdx.x;
    const int lane = tid & 31;
    const int wave = tid >> 5;
    const int n    = blockIdx.x;
    const int j0   = blockIdx.y * 128;
    const size_t baseN = (size_t)n * N_SEQ * C_DIM;

    // stage transposed f16 weights (straight u32 copy into padded rows)
    {
        const uint32_t* s1 = (const uint32_t*)WTle_g;
        const uint32_t* s2 = (const uint32_t*)WTg_g;
        for (int u = tid; u < 8192; u += 256) {
            const int row = u >> 6, col = u & 63;
            sWle[row * LDW + col] = s1[u];
            sWg [row * LDW + col] = s2[u];
        }
    }
    // stage pair chunk as f16 (coalesced global reads)
    {
        const int c = tid & 127;
        for (int r = tid >> 7; r < 128; r += 2)
            sA_h[r * LDH + c] = (_Float16)pair[baseN + (size_t)(j0 + r) * C_DIM + c];
    }
    if (tid < 128) rsL[tid] = rs_g[n * H_DIM + tid];
    __syncthreads();

    const int nl    = lane & 15;
    const int rhalf = (lane >> 4) * 8;

    // ---- GEMM1+2 fused: T = (sA@Wle + ble) * rs * sigmoid(sA@Wg + bg) -----
    {
        const int m0 = wave * 16;
        v16h a0 = load_frag_A(sA, m0,  0, lane);
        v16h a1 = load_frag_A(sA, m0, 32, lane);
        v16h a2 = load_frag_A(sA, m0, 64, lane);
        v16h a3 = load_frag_A(sA, m0, 96, lane);
#pragma unroll 1
        for (int tc = 0; tc < 8; ++tc) {
            const int h0 = tc * 16;
            const int h  = h0 + nl;
            const float blv = ble_g[h], bgv = bg_g[h];
            v8f accL, accG;
#pragma unroll
            for (int r = 0; r < 8; ++r) { accL[r] = blv; accG[r] = bgv; }
            v16h b;
            b = load_frag_B(sWle, h0,  0, lane);
            accL = __builtin_amdgcn_wmma_f32_16x16x32_f16(false, a0, false, b, (short)0, accL, false, false);
            b = load_frag_B(sWle, h0, 32, lane);
            accL = __builtin_amdgcn_wmma_f32_16x16x32_f16(false, a1, false, b, (short)0, accL, false, false);
            b = load_frag_B(sWle, h0, 64, lane);
            accL = __builtin_amdgcn_wmma_f32_16x16x32_f16(false, a2, false, b, (short)0, accL, false, false);
            b = load_frag_B(sWle, h0, 96, lane);
            accL = __builtin_amdgcn_wmma_f32_16x16x32_f16(false, a3, false, b, (short)0, accL, false, false);
            b = load_frag_B(sWg, h0,  0, lane);
            accG = __builtin_amdgcn_wmma_f32_16x16x32_f16(false, a0, false, b, (short)0, accG, false, false);
            b = load_frag_B(sWg, h0, 32, lane);
            accG = __builtin_amdgcn_wmma_f32_16x16x32_f16(false, a1, false, b, (short)0, accG, false, false);
            b = load_frag_B(sWg, h0, 64, lane);
            accG = __builtin_amdgcn_wmma_f32_16x16x32_f16(false, a2, false, b, (short)0, accG, false, false);
            b = load_frag_B(sWg, h0, 96, lane);
            accG = __builtin_amdgcn_wmma_f32_16x16x32_f16(false, a3, false, b, (short)0, accG, false, false);
            const float rv = rsL[h];
#pragma unroll
            for (int r = 0; r < 8; ++r) {
                const float g = 1.f / (1.f + __expf(-accG[r]));
                const float t = accL[r] * rv * g;
                sT_h[(m0 + r + rhalf) * LDH + h] = (_Float16)t;
            }
        }
    }
    __syncthreads();

    // restage W_out^T over dead region (W_le slot)
    {
        const uint32_t* s3 = (const uint32_t*)WTo_g;
        for (int u = tid; u < 8192; u += 256) {
            const int row = u >> 6, col = u & 63;
            sWle[row * LDW + col] = s3[u];
        }
    }
    __syncthreads();

    // ---- GEMM3 + residual + LayerNorm (all in registers) -------------------
    {
        const int m0 = wave * 16;
        v16h a0 = load_frag_A(sT, m0,  0, lane);
        v16h a1 = load_frag_A(sT, m0, 32, lane);
        v16h a2 = load_frag_A(sT, m0, 64, lane);
        v16h a3 = load_frag_A(sT, m0, 96, lane);
        float accO[8][8];
#pragma unroll
        for (int tc = 0; tc < 8; ++tc) {
            const int c0 = tc * 16;
            const int c  = c0 + nl;
            const float bov = bo_g[c];
            v8f acc;
#pragma unroll
            for (int r = 0; r < 8; ++r) acc[r] = bov;
            v16h b;
            b = load_frag_B(sWle, c0,  0, lane);
            acc = __builtin_amdgcn_wmma_f32_16x16x32_f16(false, a0, false, b, (short)0, acc, false, false);
            b = load_frag_B(sWle, c0, 32, lane);
            acc = __builtin_amdgcn_wmma_f32_16x16x32_f16(false, a1, false, b, (short)0, acc, false, false);
            b = load_frag_B(sWle, c0, 64, lane);
            acc = __builtin_amdgcn_wmma_f32_16x16x32_f16(false, a2, false, b, (short)0, acc, false, false);
            b = load_frag_B(sWle, c0, 96, lane);
            acc = __builtin_amdgcn_wmma_f32_16x16x32_f16(false, a3, false, b, (short)0, acc, false, false);
#pragma unroll
            for (int r = 0; r < 8; ++r)
                accO[tc][r] = acc[r] + pair[baseN + (size_t)(j0 + m0 + r + rhalf) * C_DIM + c];
        }
#pragma unroll
        for (int r = 0; r < 8; ++r) {
            float s = 0.f, q = 0.f;
#pragma unroll
            for (int tc = 0; tc < 8; ++tc) { const float v = accO[tc][r]; s += v; q += v * v; }
#pragma unroll
            for (int off = 8; off; off >>= 1) {    // xor<=8 stays within 16-lane row group
                s += __shfl_xor(s, off, 32);
                q += __shfl_xor(q, off, 32);
            }
            const float mu   = s * (1.f / 128.f);
            const float var  = q * (1.f / 128.f) - mu * mu;
            const float rsig = rsqrtf(var + 1e-5f);
            const size_t rowOff = baseN + (size_t)(j0 + m0 + r + rhalf) * C_DIM;
#pragma unroll
            for (int tc = 0; tc < 8; ++tc) {
                const int c = tc * 16 + nl;
                out[rowOff + c] = (accO[tc][r] - mu) * rsig * lnsc[c] + lnb[c];
            }
        }
    }
}

extern "C" void kernel_launch(void* const* d_in, const int* in_sizes, int n_in,
                              void* d_out, int out_size, void* d_ws, size_t ws_size,
                              hipStream_t stream) {
    const float* pair    = (const float*)d_in[0];
    const float* W_left  = (const float*)d_in[1];
    const float* b_left  = (const float*)d_in[2];
    const float* W_right = (const float*)d_in[3];
    const float* b_right = (const float*)d_in[4];
    const float* W_gate  = (const float*)d_in[5];
    const float* b_gate  = (const float*)d_in[6];
    const float* W_eq    = (const float*)d_in[7];
    const float* b_eq    = (const float*)d_in[8];
    const float* W_out   = (const float*)d_in[9];
    const float* b_out   = (const float*)d_in[10];
    const float* ln_sc   = (const float*)d_in[11];
    const float* ln_b    = (const float*)d_in[12];
    float* out = (float*)d_out;

    char* ws = (char*)d_ws;
    float*    rs_g  = (float*)ws;                       // 384*128*4 = 196608
    _Float16* WTle  = (_Float16*)(ws + 196608);         // 32768
    _Float16* WTg   = (_Float16*)(ws + 229376);         // 32768
    _Float16* WTo   = (_Float16*)(ws + 262144);         // 32768
    float*    b_le  = (float*)(ws + 294912);            // 512

    prep_weights<<<192, 256, 0, stream>>>(W_left, W_eq, W_gate, W_out,
                                          b_left, b_eq, WTle, WTg, WTo, b_le);
    row_sums<<<N_SEQ, 128, 0, stream>>>(pair, W_right, b_right, rs_g);

    const size_t smem = 133632;  // 3*33280 (region0) + 33280 (sT) + 512 (rs)
    tri_main<<<dim3(N_SEQ, 3), 256, smem, stream>>>(pair, rs_g, WTle, WTg, WTo,
                                                    b_le, b_gate, b_out, ln_sc, ln_b, out);
}
